// SelfAttention_6811818131513
// MI455X (gfx1250) — matrix operand order repaired and checked by reading, not compiled, not run on hardware
//
#include <hip/hip_runtime.h>

typedef __attribute__((ext_vector_type(16))) _Float16 v16h;
typedef __attribute__((ext_vector_type(8)))  _Float16 v8h;
typedef __attribute__((ext_vector_type(8)))  float    v8f;
typedef __attribute__((ext_vector_type(4)))  int      v4i;

typedef __attribute__((address_space(1))) v4i* as1_v4i_p;   // global int4*
typedef __attribute__((address_space(3))) v4i* as3_v4i_p;   // LDS int4*

constexpr int NB = 2;
constexpr int S  = 2048;
constexpr int E  = 1024;
constexpr int H  = 16;
constexpr int D  = 64;       // head dim
constexpr int CHUNK = 32;    // keys per flash-attention iteration
constexpr int WAVES = 8;     // waves per block
constexpr float SCALE = 0.03125f;   // 1/sqrt(E) = 1/32

#if __has_builtin(__builtin_amdgcn_global_load_async_to_lds_b128) && \
    __has_builtin(__builtin_amdgcn_s_wait_asynccnt)
#define USE_ASYNC_LDS 1
#else
#define USE_ASYNC_LDS 0
#endif

// ---------------------------------------------------------------------------
// Kernel 1: flash attention.  One wave = one 16-query tile of one (batch,head).
// Block = 8 waves = 128 queries of one head.  K chunk staged global->LDS via
// async-to-LDS (no VGPR transit, ASYNCcnt); V chunk staged transposed as f16
// through VGPRs (needs transpose + convert anyway).  Score slab / softmax
// bookkeeping are wave-local LDS (in-order DS within a wave), so they use a
// scheduling fence instead of block barriers.
// ---------------------------------------------------------------------------
__global__ __launch_bounds__(256) void attn_flash_kernel(
    const float* __restrict__ Vg,   // values [N,S,E]
    const float* __restrict__ Kg,   // keys   [N,S,E]
    const float* __restrict__ Qg,   // query  [N,S,E]
    const int*   __restrict__ Mg,   // mask   [N,1,S,S]
    _Float16*    __restrict__ Xout) // ws     [N*S, E] f16
{
    alignas(64) __shared__ float    sKf[CHUNK][D];  // [key][dim] f32 (async copy)
    alignas(64) __shared__ _Float16 sVt[D][CHUNK];  // [dim][key] f16 (B frags for P@V)
    __shared__ float sS[WAVES][16][33];             // per-wave score/prob slab (+pad)
    __shared__ float sAlpha[WAVES][16];             // per-row rescale / rowsum bcast

    const int lane = threadIdx.x & 31;
    const int wave = threadIdx.x >> 5;
    const int qg   = blockIdx.x;     // 0..15  (S / 128)
    const int h    = blockIdx.y;     // 0..15
    const int n    = blockIdx.z;     // 0..1
    const int qbase = qg * 128 + wave * 16;

    const int col = lane & 15;       // C-layout column / A-layout row (M)
    const int hi  = lane >> 4;       // lane half

    // ---- Q tile as two A-fragments (head-dim halves 0..31 and 32..63) ----
    v16h aq[2];
    {
        const int kb = hi ? 8 : 0;
        const float* qrow = Qg + ((size_t)(n * S + qbase + col)) * E + h * D;
#pragma unroll
        for (int dh = 0; dh < 2; ++dh) {
#pragma unroll
            for (int j = 0; j < 8; ++j)
                aq[dh][j]     = (_Float16)qrow[dh * 32 + kb + j];
#pragma unroll
            for (int j = 0; j < 8; ++j)
                aq[dh][8 + j] = (_Float16)qrow[dh * 32 + kb + 16 + j];
        }
    }

    v8f acc[4] = {{}, {}, {}, {}};   // 16q x 64d output accumulators (f32)
    float run_max = -3.0e38f;
    float run_sum = 0.0f;

    for (int kc = 0; kc < S; kc += CHUNK) {
        __syncthreads();             // previous chunk's sKf/sVt readers done
        // ---- cooperative stage of K (async, f32) and V (transposed f16) ----
        {
            const int t   = threadIdx.x;
            const int kk  = t >> 3;          // 0..31 key within chunk
            const int seg = (t & 7) * 8;     // dim base, 8 consecutive dims
            const float* krow = Kg + ((size_t)(n * S + kc + kk)) * E + h * D + seg;
            const float* vrow = Vg + ((size_t)(n * S + kc + kk)) * E + h * D + seg;
#if USE_ASYNC_LDS
            __builtin_amdgcn_global_load_async_to_lds_b128(
                (as1_v4i_p)(void*)krow,
                (as3_v4i_p)(void*)&sKf[kk][seg], 0, 0);
            __builtin_amdgcn_global_load_async_to_lds_b128(
                (as1_v4i_p)(void*)krow,
                (as3_v4i_p)(void*)&sKf[kk][seg], 16, 0);
#else
#pragma unroll
            for (int j = 0; j < 8; ++j)
                sKf[kk][seg + j] = krow[j];
#endif
#pragma unroll
            for (int j = 0; j < 8; ++j)
                sVt[seg + j][kk] = (_Float16)vrow[j];
        }
#if USE_ASYNC_LDS
        __builtin_amdgcn_s_wait_asynccnt(0);
#endif
        __syncthreads();

        // ---- S = Q K^T : two 16-key tiles, each summed over head dim ----
        {
            const int kb = hi ? 8 : 0;       // B-layout K base (head-dim here)
#pragma unroll
            for (int kt = 0; kt < 2; ++kt) {
                v8f c = {};
#pragma unroll
                for (int dh = 0; dh < 2; ++dh) {
                    const v8f b0 = *(const v8f*)&sKf[kt * 16 + col][dh * 32 + kb];
                    const v8f b1 = *(const v8f*)&sKf[kt * 16 + col][dh * 32 + kb + 16];
                    v16h bk;
#pragma unroll
                    for (int j = 0; j < 8; ++j) {
                        bk[j]     = (_Float16)b0[j];
                        bk[8 + j] = (_Float16)b1[j];
                    }
                    c = __builtin_amdgcn_wmma_f32_16x16x32_f16(
                            false, aq[dh], false, bk, (short)0, c, false, false);
                }
                // C layout: lane owns key-col n=col, rows m = i + 8*hi
#pragma unroll
                for (int i = 0; i < 8; ++i)
                    sS[wave][i + 8 * hi][kt * 16 + col] = c[i];
            }
        }
        __builtin_amdgcn_wave_barrier();   // wave-local LDS handoff (DS in-order)

        // ---- online softmax: lane handles (row = col, key half = hi) ----
        {
            const int r = col;
            const int q = qbase + r;
            const int* mrow = Mg + ((size_t)n * S + q) * S + kc + hi * 16;
            float sv[16];
            float lmax = -3.0e38f;
#pragma unroll
            for (int t = 0; t < 16; ++t) {
                float s = sS[wave][r][hi * 16 + t];
                s = (mrow[t] == 0) ? -1.0e20f : s;   // mask, then scale (as reference)
                s *= SCALE;
                sv[t] = s;
                lmax = fmaxf(lmax, s);
            }
            lmax = fmaxf(lmax, __shfl_xor(lmax, 16, 32));
            const float new_max = fmaxf(run_max, lmax);
            const float alpha   = __expf(run_max - new_max);
            float lsum = 0.0f;
#pragma unroll
            for (int t = 0; t < 16; ++t) {
                const float p = __expf(sv[t] - new_max);
                sS[wave][r][hi * 16 + t] = p;        // overwrite scores with probs
                lsum += p;
            }
            lsum += __shfl_xor(lsum, 16, 32);
            run_sum = run_sum * alpha + lsum;
            run_max = new_max;
            if (hi == 0) sAlpha[wave][r] = alpha;
        }
        __builtin_amdgcn_wave_barrier();

        // ---- rescale existing accumulators by per-row alpha ----
#pragma unroll
        for (int i = 0; i < 8; ++i) {
            const float a = sAlpha[wave][i + 8 * hi];
#pragma unroll
            for (int nd = 0; nd < 4; ++nd)
                acc[nd][i] *= a;
        }

        // ---- P (f16 A-fragment 16 x 32keys) ----
        v16h pa;
        {
            const int kb = hi ? 16 : 0;
#pragma unroll
            for (int j = 0; j < 8; ++j)
                pa[j]     = (_Float16)sS[wave][col][kb + j];
#pragma unroll
            for (int j = 0; j < 8; ++j)
                pa[8 + j] = (_Float16)sS[wave][col][kb + 8 + j];
        }
        // ---- O += P V : 4 dim-tiles of 16, K = 32 keys in one WMMA each ----
        {
            const int kb2 = hi ? 16 : 0;
#pragma unroll
            for (int nd = 0; nd < 4; ++nd) {
                const v16h bv = *(const v16h*)&sVt[nd * 16 + col][kb2];
                acc[nd] = __builtin_amdgcn_wmma_f32_16x16x32_f16(
                              false, pa, false, bv, (short)0, acc[nd], false, false);
            }
        }
    }

    // ---- normalize by row sum and write f16 to workspace ----
    __builtin_amdgcn_wave_barrier();
    if (hi == 0) sAlpha[wave][col] = run_sum;
    __builtin_amdgcn_wave_barrier();
#pragma unroll
    for (int i = 0; i < 8; ++i) {
        const int m = i + 8 * hi;
        const float inv = 1.0f / sAlpha[wave][m];
        const size_t idx = ((size_t)(n * S + qbase + m)) * E + h * D;
#pragma unroll
        for (int nd = 0; nd < 4; ++nd)
            Xout[idx + nd * 16 + col] = (_Float16)(acc[nd][i] * inv);
    }
}

// ---------------------------------------------------------------------------
// Kernel 2: out = x @ W^T + b.  M=4096, N=1024, K=1024.  One 16x64 output
// strip per wave (4 accumulators): each A fragment is reused by 4 WMMAs.
// B-fragment reads are 16 contiguous floats of a W row (K contiguous for
// x @ W^T), converted to f16 on load; prefetch covers the cold first pass.
// ---------------------------------------------------------------------------
__global__ __launch_bounds__(256) void proj_gemm_kernel(
    const _Float16* __restrict__ X,   // [N*S, E] f16
    const float*    __restrict__ W,   // [E, E]
    const float*    __restrict__ bvec,// [E]
    float*          __restrict__ Out) // [N*S, E]
{
    const int lane = threadIdx.x & 31;
    const int wave = threadIdx.x >> 5;
    const int task = blockIdx.x * WAVES + wave;   // 4096 strips
    const int rm   = task >> 4;                   // 256 row tiles
    const int cg   = task & 15;                   // 16 col groups of 64
    const int col  = lane & 15;
    const int hi   = lane >> 4;

    const size_t arow = (size_t)(rm * 16 + col) * E;
    const int kbA = hi ? 16 : 0;
    const int kbB = hi ? 16 : 0;

    v8f acc[4] = {{}, {}, {}, {}};
    for (int kk = 0; kk < E; kk += 32) {
        const v8h a0 = *(const v8h*)(X + arow + kk + kbA);
        const v8h a1 = *(const v8h*)(X + arow + kk + kbA + 8);
        const v16h a = __builtin_shufflevector(a0, a1,
                           0, 1, 2, 3, 4, 5, 6, 7, 8, 9, 10, 11, 12, 13, 14, 15);
#pragma unroll
        for (int nt = 0; nt < 4; ++nt) {
            const int e = cg * 64 + nt * 16 + col;
            const float* wrow = W + (size_t)e * E + kk + kbB;
            if (kk + 32 < E)
                __builtin_prefetch(wrow + 32, 0, 1);   // next K chunk of this row
            v16h bf;
#pragma unroll
            for (int j = 0; j < 16; ++j) bf[j] = (_Float16)wrow[j];
            acc[nt] = __builtin_amdgcn_wmma_f32_16x16x32_f16(
                          false, a, false, bf, (short)0, acc[nt], false, false);
        }
    }

#pragma unroll
    for (int nt = 0; nt < 4; ++nt) {
        const int e = cg * 64 + nt * 16 + col;
        const float bias = bvec[e];
#pragma unroll
        for (int i = 0; i < 8; ++i) {
            const int m = i + 8 * hi;
            Out[(size_t)(rm * 16 + m) * E + e] = acc[nt][i] + bias;
        }
    }
}

extern "C" void kernel_launch(void* const* d_in, const int* in_sizes, int n_in,
                              void* d_out, int out_size, void* d_ws, size_t ws_size,
                              hipStream_t stream) {
    const float* Vg = (const float*)d_in[0];   // values
    const float* Kg = (const float*)d_in[1];   // keys
    const float* Qg = (const float*)d_in[2];   // query
    const int*   Mg = (const int*)  d_in[3];   // mask
    const float* Wo = (const float*)d_in[4];   // W_out
    const float* bo = (const float*)d_in[5];   // b_out
    float* Out = (float*)d_out;
    _Float16* X = (_Float16*)d_ws;             // 8 MB attention output, f16

    dim3 grid1(S / 128, H, NB);                // 16 x 16 x 2 blocks
    attn_flash_kernel<<<grid1, 256, 0, stream>>>(Vg, Kg, Qg, Mg, X);

    const int strips = (NB * S / 16) * (E / 64);   // 4096
    proj_gemm_kernel<<<strips / WAVES, 256, 0, stream>>>(X, Wo, bo, Out);
}
